// ChineseCLIPVisionLayer_55336358643322
// MI455X (gfx1250) — matrix operand order repaired and checked by reading, not compiled, not run on hardware
//
#include <hip/hip_runtime.h>

// ---------------- problem constants ----------------
#define BB   32
#define SS   257
#define DD   1024
#define HH   16
#define HDD  64
#define FFD  4096
#define MTOK (BB * SS)          // 8224 tokens
#define QSCALE 0.125f           // 64^-0.5

typedef unsigned short u16;
typedef unsigned int   u32;
typedef unsigned long long u64;

typedef __attribute__((ext_vector_type(16))) __bf16 v16bf;
typedef __attribute__((ext_vector_type(8)))  float  v8f;
typedef __attribute__((ext_vector_type(4)))  unsigned int v4u;
typedef __attribute__((ext_vector_type(8)))  int v8i;
typedef __attribute__((ext_vector_type(4)))  int v4i;

union FragU  { uint4 q[2]; v16bf v; };   // 32B fragment built from two 16B loads
union Frag16 { u16   s[16]; v16bf v; };  // 32B fragment built element-wise

// fp32 -> bf16, round-to-nearest-even
__device__ __forceinline__ u16 f2bf(float f) {
  u32 u = __float_as_uint(f);
  u += 0x7fffu + ((u >> 16) & 1u);
  return (u16)(u >> 16);
}

__device__ __forceinline__ v8f wmma_bf16(v16bf a, v16bf b, v8f c) {
  // v_wmma_f32_16x16x32_bf16  D = A*B + C, f32 accumulate
  return __builtin_amdgcn_wmma_f32_16x16x32_bf16(false, a, false, b, (short)0, c, false, false);
}

// reductions within a 16-lane group (matches C-fragment row distribution on wave32)
__device__ __forceinline__ float redmax16(float v) {
  v = fmaxf(v, __shfl_xor(v, 1, 32));
  v = fmaxf(v, __shfl_xor(v, 2, 32));
  v = fmaxf(v, __shfl_xor(v, 4, 32));
  v = fmaxf(v, __shfl_xor(v, 8, 32));
  return v;
}
__device__ __forceinline__ float redsum16(float v) {
  v += __shfl_xor(v, 1, 32);
  v += __shfl_xor(v, 2, 32);
  v += __shfl_xor(v, 4, 32);
  v += __shfl_xor(v, 8, 32);
  return v;
}

// ---------------- Tensor Data Mover: DMA a [tile_rows x 32] bf16 tile into LDS ----------------
// Hardware applies LDS padding of 4 dwords (16B) every 16 dwords (64B = one 32-elem row),
// reproducing the stride-40 (bf16 elems) bank-spread layout. Rows >= rows_avail zero-fill (OOB).
__device__ __forceinline__ void tdm_load_tile(const u16* gsrc, u32 lds_byte_off,
                                              u32 rows_avail, u32 tile_rows,
                                              u32 row_stride_elems) {
  u64 ga = (u64)(size_t)gsrc;
  v4u g0;
  g0.x = 1u;                                  // count=1, no gather
  g0.y = lds_byte_off;                        // LDS destination (bytes)
  g0.z = (u32)ga;                             // global_addr[31:0]
  g0.w = (u32)(ga >> 32) | (2u << 30);        // global_addr[56:32] | type=2 ("image")
  u64 s0 = (u64)row_stride_elems;             // tensor_dim0_stride (data_size units)
  u32 td0 = row_stride_elems;                 // tensor_dim0: full line, no X OOB
  u32 td1 = rows_avail;                       // tensor_dim1: rows valid -> Y OOB zero-fill
  v8i g1;
  g1[0] = (int)((1u << 16)        // data_size = 1 -> 2 bytes
              | (1u << 20)        // pad_enable
              | (3u << 22)        // pad_interval: 16 dwords (one 64B row)
              | (3u << 25));      // pad_amount: 4 dwords (16B)
  g1[1] = (int)((td0 & 0xFFFFu) << 16);                    // tensor_dim0[15:0]
  g1[2] = (int)((td0 >> 16) | ((td1 & 0xFFFFu) << 16));    // dim0[31:16] | dim1[15:0]
  g1[3] = (int)((td1 >> 16) | (32u << 16));                // dim1[31:16] | tile_dim0=32
  g1[4] = (int)(tile_rows & 0xFFFFu);                      // tile_dim1 (tile_dim2=0)
  g1[5] = (int)(u32)(s0 & 0xFFFFFFFFull);                  // stride0[31:0]
  g1[6] = (int)(u32)((s0 >> 32) & 0xFFFFull);              // stride0[47:32] | stride1 lo = 0
  g1[7] = 0;
  v4i z4 = {0, 0, 0, 0};
#if defined(__clang_major__) && (__clang_major__ >= 23)
  v8i z8 = {0, 0, 0, 0, 0, 0, 0, 0};
  __builtin_amdgcn_tensor_load_to_lds(g0, g1, z4, z4, z8, 0);
#else
  __builtin_amdgcn_tensor_load_to_lds(g0, g1, z4, z4, 0);
#endif
}

// ---------------- fp32 -> bf16 convert ----------------
__global__ __launch_bounds__(256) void cvt_bf16(const float* __restrict__ in,
                                                u16* __restrict__ out, int n) {
  int i = blockIdx.x * 256 + threadIdx.x;
  if (i < n) out[i] = f2bf(in[i]);
}

// ---------------- LayerNorm (fp32 in, bf16 out), one block per token ----------------
__global__ __launch_bounds__(256) void ln_kernel(const float* __restrict__ x,
                                                 const float* __restrict__ g,
                                                 const float* __restrict__ bta,
                                                 u16* __restrict__ out) {
  __shared__ float r1[256];
  __shared__ float r2[256];
  const int row = blockIdx.x;
  const int tid = threadIdx.x;
  const float* xp = x + (size_t)row * DD;

  float lv[4];
  float s = 0.f, s2 = 0.f;
  #pragma unroll
  for (int i = 0; i < 4; i++) {
    float vv = xp[tid + i * 256];
    lv[i] = vv; s += vv; s2 += vv * vv;
  }
  r1[tid] = s; r2[tid] = s2;
  __syncthreads();
  for (int st = 128; st > 0; st >>= 1) {
    if (tid < st) { r1[tid] += r1[tid + st]; r2[tid] += r2[tid + st]; }
    __syncthreads();
  }
  float mu  = r1[0] * (1.0f / DD);
  float var = r2[0] * (1.0f / DD) - mu * mu;
  float rs  = rsqrtf(var + 1e-5f);
  #pragma unroll
  for (int i = 0; i < 4; i++) {
    int c = tid + i * 256;
    out[(size_t)row * DD + c] = f2bf((lv[i] - mu) * rs * g[c] + bta[c]);
  }
}

// ---------------- WMMA bf16 GEMM with TDM-staged, double-buffered LDS tiles ----------------
//  out[M,N] = act((A[M,K] @ W[N,K]^T + bias) * alpha) + resid
//  outmode: 0 -> fp32 [M,N], 1 -> bf16 [M,N], 2 -> bf16 scattered to [B,H,S,HD]
#define LDT  40                 // padded LDS row stride (bf16 elems) for a 32-wide K tile
#define BM   64
#define BN   256
#define ABUF (BM * LDT)         // 2560 elems per buffer
#define WBUF (BN * LDT)         // 10240 elems per buffer

__global__ __launch_bounds__(256) void gemm_bf16_wmma(
    const u16* __restrict__ A, const u16* __restrict__ W,
    const float* __restrict__ bias, const float* __restrict__ resid,
    float* __restrict__ outf, u16* __restrict__ outb,
    int M, int N, int K, float alpha, int act, int outmode)
{
  __shared__ u16 As[2 * ABUF];   // 2 x (64 x 32) A tiles, hw-padded stride 40
  __shared__ u16 Ws[2 * WBUF];   // 2 x (256 x 32) W tiles

  const int t   = threadIdx.x;
  const int l   = t & 31;
  const int lg  = l & 15;
  const int lh  = l >> 4;
  const int wid = t >> 5;                 // 8 waves: 2 (M) x 4 (N)
  const int wm  = (wid >> 2) * 32;
  const int wn  = (wid & 3) * 64;
  const int m0  = blockIdx.x * BM;
  const int n0  = blockIdx.y * BN;

  const u32 asBase = (u32)(size_t)(&As[0]);
  const u32 wsBase = (u32)(size_t)(&Ws[0]);
  const u32 rowsA  = (u32)(M > m0 ? M - m0 : 0);

  v8f acc[2][4];
  #pragma unroll
  for (int i = 0; i < 2; i++)
    #pragma unroll
    for (int j = 0; j < 4; j++)
      #pragma unroll
      for (int r = 0; r < 8; r++) acc[i][j][r] = 0.0f;

  const int steps = K >> 5;

  // prologue: DMA tile 0 into buffer 0 (single wave issues; TENSORcnt tracks)
  if (wid == 0) {
    tdm_load_tile(A + (size_t)m0 * K, asBase, rowsA, BM, (u32)K);
    tdm_load_tile(W + (size_t)n0 * K, wsBase, (u32)(N - n0), BN, (u32)K);
  }

  for (int s = 0; s < steps; s++) {
    const int buf = s & 1;
    if (wid == 0) __builtin_amdgcn_s_wait_tensorcnt(0);
    __syncthreads();                       // tile s resident & everyone past tile s-1

    if (wid == 0 && (s + 1) < steps) {     // DMA tile s+1 into the other buffer
      const int k1 = (s + 1) << 5;
      tdm_load_tile(A + (size_t)m0 * K + k1, asBase + (u32)((buf ^ 1) * ABUF * 2),
                    rowsA, BM, (u32)K);
      tdm_load_tile(W + (size_t)n0 * K + k1, wsBase + (u32)((buf ^ 1) * WBUF * 2),
                    (u32)(N - n0), BN, (u32)K);
    }

    const u16* Ab = As + buf * ABUF;
    const u16* Wb = Ws + buf * WBUF;

    // A fragments: lane lg = row, lane-half picks K 0..15 / 16..31 (contiguous)
    FragU fa[2];
    #pragma unroll
    for (int i = 0; i < 2; i++) {
      const u16* ap = Ab + (wm + i * 16 + lg) * LDT;
      fa[i].q[0] = *reinterpret_cast<const uint4*>(ap + lh * 16);
      fa[i].q[1] = *reinterpret_cast<const uint4*>(ap + lh * 16 + 8);
    }
    // B fragments: lane lg = column n, lane-half picks K 0..15 / 16..31 (contiguous)
    FragU fb[4];
    #pragma unroll
    for (int j = 0; j < 4; j++) {
      const u16* bp = Wb + (wn + j * 16 + lg) * LDT + lh * 16;
      fb[j].q[0] = *reinterpret_cast<const uint4*>(bp);
      fb[j].q[1] = *reinterpret_cast<const uint4*>(bp + 8);
    }
    #pragma unroll
    for (int i = 0; i < 2; i++)
      #pragma unroll
      for (int j = 0; j < 4; j++)
        acc[i][j] = wmma_bf16(fa[i].v, fb[j].v, acc[i][j]);

    __syncthreads();                       // compute done before buf is re-DMA'd
  }

  // epilogue; C-frag: VGPR r -> row (lh*8 + r), lane lg -> column
  #pragma unroll
  for (int i = 0; i < 2; i++) {
    #pragma unroll
    for (int j = 0; j < 4; j++) {
      int n = n0 + wn + j * 16 + lg;
      float bs = bias[n];
      #pragma unroll
      for (int r = 0; r < 8; r++) {
        int m = m0 + wm + i * 16 + lh * 8 + r;
        if (m >= M) continue;
        float vv = (acc[i][j][r] + bs) * alpha;
        if (act == 1) vv = vv / (1.0f + __expf(-1.702f * vv));   // quick-GELU
        if (resid) vv += resid[(size_t)m * N + n];
        if (outmode == 0) {
          outf[(size_t)m * N + n] = vv;
        } else if (outmode == 1) {
          outb[(size_t)m * N + n] = f2bf(vv);
        } else { // scatter [B*S, D] -> [B, H, S, HD] bf16
          int b = m / SS, ss = m - b * SS;
          int hh = n >> 6, hd = n & 63;
          outb[(((size_t)(b * HH + hh)) * SS + ss) * HDD + hd] = f2bf(vv);
        }
      }
    }
  }
}

// ---------------- fused flash attention: one wave per (b, h, 16-query tile) ----------------
#define VLD 66   // padded V LDS stride (keys 32 x hd 64)
#define PLD 40   // padded P LDS stride (16 x 32)

__global__ __launch_bounds__(32) void attn_fused(
    const u16* __restrict__ q, const u16* __restrict__ k, const u16* __restrict__ v,
    u16* __restrict__ out)
{
  __shared__ u16 Ps[16 * PLD];
  __shared__ u16 Vs[32 * VLD];

  const int QT = (SS + 15) / 16;            // 17 query tiles
  const int qt = blockIdx.x % QT;
  const int bh = blockIdx.x / QT;
  const int b  = bh / HH;
  const int hh = bh - b * HH;

  const int l  = threadIdx.x;
  const int lg = l & 15;
  const int lh = l >> 4;

  const size_t base = (size_t)bh * SS * HDD;  // q/k/v are [B,H,S,HD] bf16

  // Q A-fragments (hd 0..31 and 32..63), rows clamped for the padded tail tile
  int qrow = qt * 16 + lg; if (qrow >= SS) qrow = SS - 1;
  const u16* qp = q + base + (size_t)qrow * HDD;
  FragU qa[2];
  #pragma unroll
  for (int hs = 0; hs < 2; hs++) {
    qa[hs].q[0] = *reinterpret_cast<const uint4*>(qp + hs * 32 + lh * 16);
    qa[hs].q[1] = *reinterpret_cast<const uint4*>(qp + hs * 32 + lh * 16 + 8);
  }

  float mrow[8], lrow[8], scl[8];
  v8f o[4];
  #pragma unroll
  for (int r = 0; r < 8; r++) { mrow[r] = -3.0e38f; lrow[r] = 0.0f; }
  #pragma unroll
  for (int tt = 0; tt < 4; tt++)
    #pragma unroll
    for (int r = 0; r < 8; r++) o[tt][r] = 0.0f;

  const int KT = (SS + 31) / 32;            // 9 key tiles of 32
  for (int kt = 0; kt < KT; kt++) {
    const int kb = kt * 32;

    { // stage one V key-row per lane (clamped; masked keys get P==0 anyway)
      int vr = kb + l; if (vr >= SS) vr = SS - 1;
      const uint4* vp = reinterpret_cast<const uint4*>(v + base + (size_t)vr * HDD);
      #pragma unroll
      for (int c = 0; c < 8; c++) {
        uint4 d = vp[c];
        u32* dst = reinterpret_cast<u32*>(Vs + l * VLD + c * 8);
        dst[0] = d.x; dst[1] = d.y; dst[2] = d.z; dst[3] = d.w;
      }
    }
    if (kt + 1 < KT) { // prefetch next K/V tiles
      int nv = kb + 32 + l; if (nv >= SS) nv = SS - 1;
      __builtin_prefetch(v + base + (size_t)nv * HDD, 0, 1);
      __builtin_prefetch(k + base + (size_t)nv * HDD, 0, 1);
    }
    __syncthreads();

    // scores: two 16x16 C-frags (key halves), each = 2 WMMAs over hd
    v8f sc[2];
    #pragma unroll
    for (int f = 0; f < 2; f++) {
      v8f c;
      #pragma unroll
      for (int r = 0; r < 8; r++) c[r] = 0.0f;
      int krow = kb + f * 16 + lg;
      int krc  = krow < SS ? krow : SS - 1;
      #pragma unroll
      for (int hs = 0; hs < 2; hs++) {
        const u16* kp = k + base + (size_t)krc * HDD + hs * 32 + lh * 16;
        FragU fbq;
        fbq.q[0] = *reinterpret_cast<const uint4*>(kp);
        fbq.q[1] = *reinterpret_cast<const uint4*>(kp + 8);
        c = wmma_bf16(qa[hs].v, fbq.v, c);
      }
      if (krow >= SS) {
        #pragma unroll
        for (int r = 0; r < 8; r++) c[r] = -1.0e30f;
      }
      sc[f] = c;
    }

    // online softmax (row stats live replicated across each 16-lane group)
    #pragma unroll
    for (int r = 0; r < 8; r++) {
      float mx = redmax16(fmaxf(sc[0][r], sc[1][r]));
      float mn = fmaxf(mrow[r], mx);
      float e0 = __expf(sc[0][r] - mn);
      float e1 = __expf(sc[1][r] - mn);
      float rs = redsum16(e0 + e1);
      scl[r]  = __expf(mrow[r] - mn);
      lrow[r] = lrow[r] * scl[r] + rs;
      mrow[r] = mn;
      sc[0][r] = e0; sc[1][r] = e1;
    }
    #pragma unroll
    for (int tt = 0; tt < 4; tt++)
      #pragma unroll
      for (int r = 0; r < 8; r++) o[tt][r] *= scl[r];

    // C-layout P -> LDS (bf16 16x32) -> reload in A-layout
    #pragma unroll
    for (int r = 0; r < 8; r++) {
      int pm = lh * 8 + r;
      Ps[pm * PLD + lg]      = f2bf(sc[0][r]);
      Ps[pm * PLD + 16 + lg] = f2bf(sc[1][r]);
    }
    __syncthreads();

    FragU pf;
    const u16* pp = Ps + lg * PLD;
    pf.q[0] = *reinterpret_cast<const uint4*>(pp + lh * 16);
    pf.q[1] = *reinterpret_cast<const uint4*>(pp + lh * 16 + 8);

    // O += P @ V  (B-frag gathered from V LDS: lane = hd column, K = key index)
    #pragma unroll
    for (int tt = 0; tt < 4; tt++) {
      Frag16 vb;
      int hd = tt * 16 + lg;
      #pragma unroll
      for (int e = 0; e < 16; e++)
        vb.s[e] = Vs[(lh * 16 + e) * VLD + hd];
      o[tt] = wmma_bf16(pf.v, vb.v, o[tt]);
    }
    __syncthreads();
  }

  // normalize and write attn output as bf16 [B, S, D] at column h*64 + hd
  #pragma unroll
  for (int r = 0; r < 8; r++) {
    int srow = qt * 16 + lh * 8 + r;
    if (srow >= SS) continue;
    float inv = 1.0f / lrow[r];
    size_t rowoff = ((size_t)(b * SS + srow)) * DD + hh * HDD;
    #pragma unroll
    for (int tt = 0; tt < 4; tt++)
      out[rowoff + tt * 16 + lg] = f2bf(o[tt][r] * inv);
  }
}

// ---------------- host orchestration ----------------
extern "C" void kernel_launch(void* const* d_in, const int* in_sizes, int n_in,
                              void* d_out, int out_size, void* d_ws, size_t ws_size,
                              hipStream_t stream)
{
  (void)in_sizes; (void)n_in; (void)out_size; (void)ws_size;

  const float* x    = (const float*)d_in[0];
  const float* ln1g = (const float*)d_in[1];
  const float* ln1b = (const float*)d_in[2];
  const float* ln2g = (const float*)d_in[3];
  const float* ln2b = (const float*)d_in[4];
  const float* qw   = (const float*)d_in[5];
  const float* qb   = (const float*)d_in[6];
  const float* kw   = (const float*)d_in[7];
  const float* kb   = (const float*)d_in[8];
  const float* vw   = (const float*)d_in[9];
  const float* vb   = (const float*)d_in[10];
  const float* ow   = (const float*)d_in[11];
  const float* ob   = (const float*)d_in[12];
  const float* w1   = (const float*)d_in[13];
  const float* b1   = (const float*)d_in[14];
  const float* w2   = (const float*)d_in[15];
  const float* b2   = (const float*)d_in[16];

  char*  ws  = (char*)d_ws;
  size_t off = 0;
  auto alloc = [&](size_t bytes) -> char* {
    char* p = ws + off;
    off += (bytes + 255) & ~(size_t)255;
    return p;
  };

  u16*   wq16 = (u16*)alloc((size_t)DD * DD * 2);
  u16*   wk16 = (u16*)alloc((size_t)DD * DD * 2);
  u16*   wv16 = (u16*)alloc((size_t)DD * DD * 2);
  u16*   wo16 = (u16*)alloc((size_t)DD * DD * 2);
  u16*   w116 = (u16*)alloc((size_t)FFD * DD * 2);
  u16*   w216 = (u16*)alloc((size_t)FFD * DD * 2);
  u16*   h16   = (u16*)alloc((size_t)MTOK * DD * 2);
  u16*   q16   = (u16*)alloc((size_t)MTOK * DD * 2);
  u16*   k16   = (u16*)alloc((size_t)MTOK * DD * 2);
  u16*   v16   = (u16*)alloc((size_t)MTOK * DD * 2);
  u16*   attn16= (u16*)alloc((size_t)MTOK * DD * 2);
  float* x1    = (float*)alloc((size_t)MTOK * DD * 4);
  u16*   h216  = (u16*)alloc((size_t)MTOK * DD * 2);
  u16*   ff16  = (u16*)alloc((size_t)MTOK * FFD * 2);

  // weights -> bf16
  {
    int nD = DD * DD, nF = FFD * DD;
    cvt_bf16<<<(nD + 255) / 256, 256, 0, stream>>>(qw, wq16, nD);
    cvt_bf16<<<(nD + 255) / 256, 256, 0, stream>>>(kw, wk16, nD);
    cvt_bf16<<<(nD + 255) / 256, 256, 0, stream>>>(vw, wv16, nD);
    cvt_bf16<<<(nD + 255) / 256, 256, 0, stream>>>(ow, wo16, nD);
    cvt_bf16<<<(nF + 255) / 256, 256, 0, stream>>>(w1, w116, nF);
    cvt_bf16<<<(nF + 255) / 256, 256, 0, stream>>>(w2, w216, nF);
  }

  dim3 gD((MTOK + BM - 1) / BM, DD / BN);    // (129, 4)
  dim3 gF((MTOK + BM - 1) / BM, FFD / BN);   // (129, 16)

  // LN1
  ln_kernel<<<MTOK, 256, 0, stream>>>(x, ln1g, ln1b, h16);
  // Q/K/V projections -> bf16 [B,H,S,HD]; Q scaled by 1/sqrt(hd)
  gemm_bf16_wmma<<<gD, 256, 0, stream>>>(h16, wq16, qb, nullptr, nullptr, q16,
                                         MTOK, DD, DD, QSCALE, 0, 2);
  gemm_bf16_wmma<<<gD, 256, 0, stream>>>(h16, wk16, kb, nullptr, nullptr, k16,
                                         MTOK, DD, DD, 1.0f, 0, 2);
  gemm_bf16_wmma<<<gD, 256, 0, stream>>>(h16, wv16, vb, nullptr, nullptr, v16,
                                         MTOK, DD, DD, 1.0f, 0, 2);
  // fused attention -> bf16 [B,S,D]
  attn_fused<<<BB * HH * ((SS + 15) / 16), 32, 0, stream>>>(q16, k16, v16, attn16);
  // O projection + residual(x) -> fp32 x1
  gemm_bf16_wmma<<<gD, 256, 0, stream>>>(attn16, wo16, ob, x, x1, nullptr,
                                         MTOK, DD, DD, 1.0f, 0, 0);
  // LN2
  ln_kernel<<<MTOK, 256, 0, stream>>>(x1, ln2g, ln2b, h216);
  // FC1 + quickGELU -> bf16
  gemm_bf16_wmma<<<gF, 256, 0, stream>>>(h216, w116, b1, nullptr, nullptr, ff16,
                                         MTOK, FFD, DD, 1.0f, 1, 1);
  // FC2 + residual(x1) -> fp32 output
  gemm_bf16_wmma<<<gD, 256, 0, stream>>>(ff16, w216, b2, x1, (float*)d_out, nullptr,
                                         MTOK, DD, FFD, 1.0f, 0, 0);
}
